// MultiHeadAttention_72799695667275
// MI455X (gfx1250) — matrix operand order repaired and checked by reading, not compiled, not run on hardware
//
#include <hip/hip_runtime.h>

// ---------------------------------------------------------------------------
// MI455X (gfx1250) multi-head attention forward.
//   bf16 WMMA (v_wmma_f32_16x16x32_bf16) for all four GEMM stages.
//   Attention computed transposed (S^T = K Q^T, O^T = V^T P^T): softmax
//   reduces in-lane + one shfl_xor(16); K fragments register-double-buffered;
//   per-wave LDS relayout fenced at wavefront scope (no workgroup barriers).
// ---------------------------------------------------------------------------

typedef __attribute__((ext_vector_type(16))) __bf16 v16bf;
typedef __attribute__((ext_vector_type(8)))  __bf16 v8bf;
typedef __attribute__((ext_vector_type(8)))  float  v8f;

static constexpr int DIMC = 1024;
static constexpr int SEQ  = 2048;
static constexpr int NB   = 2;
static constexpr int NH   = 16;
static constexpr int HD   = 64;
static constexpr int MTOT = NB * SEQ;   // 4096 rows total

#define WMMA_BF16(a, b, c) \
  __builtin_amdgcn_wmma_f32_16x16x32_bf16(false, (a), false, (b), (short)0, (c), false, false)

__device__ __forceinline__ v16bf frag_cat(v8bf lo, v8bf hi) {
  return __builtin_shufflevector(lo, hi, 0, 1, 2, 3, 4, 5, 6, 7, 8, 9, 10, 11, 12, 13, 14, 15);
}
// B-fragment: chunks [p..p+7] and [p+16..p+23] (column n fixed; caller adds lhi*8 for lanes 16..31)
__device__ __forceinline__ v16bf load_bfrag(const __bf16* p) {
  return frag_cat(*(const v8bf*)p, *(const v8bf*)(p + 16));
}
// A-fragment: chunks [p..p+7] and [p+16..p+23] (caller adds lhi*8 for lanes 16..31)
__device__ __forceinline__ v16bf load_afrag_bf16(const __bf16* p) {
  return frag_cat(*(const v8bf*)p, *(const v8bf*)(p + 16));
}
__device__ __forceinline__ v8bf cvt8(const float* p) {
  v8bf r;
#pragma unroll
  for (int i = 0; i < 8; ++i) r[i] = (__bf16)p[i];
  return r;
}
// wave-local LDS ordering: DS ops of one wave are in-order in HW; we only need
// to stop the compiler from moving LDS ops across this point.
__device__ __forceinline__ void wave_lds_fence() {
  __builtin_amdgcn_fence(__ATOMIC_ACQ_REL, "wavefront");
  __builtin_amdgcn_wave_barrier();
}

// ---------------------------------------------------------------------------
// Projection GEMM: Y[4096,1024] = X[4096,1024] @ W[1024,1024]^T + bias
// OMODE 0: Y -> bf16 row-major          (Q, K projections)
// OMODE 1: Y -> bf16 transposed [n][m]  (V projection -> Vt)
// OMODE 2: Y -> f32 row-major           (output projection -> d_out)
// Macro tile 256(M) x 64(N); each wave owns 32x64 (two row-blocks sharing the
// LDS B-fragments -> 16 back-to-back WMMAs per 64-wide K stage).
// ---------------------------------------------------------------------------
template <typename InT, int OMODE>
__global__ __launch_bounds__(256) void proj_kernel(const InT* __restrict__ X,
                                                   const float* __restrict__ W,
                                                   const float* __restrict__ bias,
                                                   void* __restrict__ Y) {
  __shared__ __align__(16) __bf16 lds_w[64][64];   // [n-tile][k-stage] bf16
  const int tid  = threadIdx.x;
  const int wave = tid >> 5;
  const int lane = tid & 31;
  const int lrow = lane & 15;
  const int lhi  = lane >> 4;
  const int m0   = blockIdx.x * 256 + wave * 32;
  const int n0   = blockIdx.y * 64;

  v8f acc[2][4] = {};

  // cooperative W-stage load: each thread converts 16 contiguous fp32 -> bf16
  const int e_n = tid >> 2;            // 0..63
  const int e_k = (tid & 3) * 16;      // 0,16,32,48
  const float* wbase = W + (size_t)(n0 + e_n) * DIMC + e_k;

  for (int k0 = 0; k0 < DIMC; k0 += 64) {
    *(v8bf*)&lds_w[e_n][e_k]     = cvt8(wbase + k0);
    *(v8bf*)&lds_w[e_n][e_k + 8] = cvt8(wbase + k0 + 8);
    __syncthreads();

    v16bf a[2][2];
#pragma unroll
    for (int rb = 0; rb < 2; ++rb) {
      if constexpr (sizeof(InT) == 4) {
        const float* xp = (const float*)X + (size_t)(m0 + rb * 16 + lrow) * DIMC + k0 + lhi * 8;
        a[rb][0] = frag_cat(cvt8(xp), cvt8(xp + 16));
        a[rb][1] = frag_cat(cvt8(xp + 32), cvt8(xp + 48));
      } else {
        const __bf16* xp = (const __bf16*)X + (size_t)(m0 + rb * 16 + lrow) * DIMC + k0 + lhi * 8;
        a[rb][0] = load_afrag_bf16(xp);
        a[rb][1] = load_afrag_bf16(xp + 32);
      }
    }

    v16bf b0[4], b1[4];
#pragma unroll
    for (int t = 0; t < 4; ++t) b0[t] = load_bfrag(&lds_w[t * 16 + lrow][lhi * 8]);
#pragma unroll
    for (int t = 0; t < 4; ++t) b1[t] = load_bfrag(&lds_w[t * 16 + lrow][32 + lhi * 8]);

#pragma unroll
    for (int rb = 0; rb < 2; ++rb)
#pragma unroll
      for (int t = 0; t < 4; ++t) acc[rb][t] = WMMA_BF16(a[rb][0], b0[t], acc[rb][t]);
#pragma unroll
    for (int rb = 0; rb < 2; ++rb)
#pragma unroll
      for (int t = 0; t < 4; ++t) acc[rb][t] = WMMA_BF16(a[rb][1], b1[t], acc[rb][t]);
    __syncthreads();
  }

#pragma unroll
  for (int rb = 0; rb < 2; ++rb) {
    const int mr = m0 + rb * 16 + lhi * 8;          // C layout: rows mr..mr+7 this lane
#pragma unroll
    for (int t = 0; t < 4; ++t) {
      const int n  = n0 + t * 16 + lrow;
      const float bv = bias[n];
      if constexpr (OMODE == 2) {
        float* out = (float*)Y;
#pragma unroll
        for (int i = 0; i < 8; ++i) out[(size_t)(mr + i) * DIMC + n] = acc[rb][t][i] + bv;
      } else if constexpr (OMODE == 0) {
        __bf16* out = (__bf16*)Y;
#pragma unroll
        for (int i = 0; i < 8; ++i)
          out[(size_t)(mr + i) * DIMC + n] = (__bf16)(acc[rb][t][i] + bv);
      } else {                                      // transposed: out[n][m], contiguous in m
        __bf16* out = (__bf16*)Y;
        v8bf tv;
#pragma unroll
        for (int i = 0; i < 8; ++i) tv[i] = (__bf16)(acc[rb][t][i] + bv);
        *(v8bf*)(out + (size_t)n * MTOT + mr) = tv;
      }
    }
  }
}

// ---------------------------------------------------------------------------
// Attention (transposed): per WG one (b,h) and 128 queries (8 waves x 16).
//   S^T[32,16] = K_blk[32,64] @ Q[16,64]^T   (K rows are A-frags, Q rows are B-frags)
//   softmax over keys = in-lane reduce + shfl_xor(16); m,l scalar per lane
//   O^T[64,16] += V^T[64,32] @ P^T[32,16]    (Vt rows are A-frags, P^T via LDS B-frag)
// K fragments double-buffered in registers; V fragments loaded at loop top.
// ---------------------------------------------------------------------------
__global__ __launch_bounds__(256) void attn_kernel(const __bf16* __restrict__ Qp,
                                                   const __bf16* __restrict__ Kp,
                                                   const __bf16* __restrict__ Vt,
                                                   __bf16* __restrict__ Ctx) {
  __shared__ __align__(16) __bf16 lds_pt[8][16][32];   // [wave][query][key_local]
  const int tid  = threadIdx.x;
  const int wave = tid >> 5;
  const int lane = tid & 31;
  const int lrow = lane & 15;
  const int lhi  = lane >> 4;
  const int b    = blockIdx.y >> 4;
  const int h    = blockIdx.y & 15;
  const int q0   = blockIdx.x * 128 + wave * 16;

  // Q B-fragments: column q = q0+lrow, d chunks ks*32 + lhi*8 .. +7 and +16 .. +23
  const __bf16* qp = Qp + (size_t)(b * SEQ + q0 + lrow) * DIMC + h * HD;
  const v16bf qb0 = load_bfrag(qp + lhi * 8);         // d 0..31
  const v16bf qb1 = load_bfrag(qp + 32 + lhi * 8);    // d 32..63

  v8f o[4] = {};                 // O^T tiles: rows d = t*16.., cols = queries
  float m = -1e30f, l = 0.f;     // per-lane scalars (one query column per lane)

  const __bf16* kbase = Kp + (size_t)(b * SEQ) * DIMC + h * HD + lhi * 8;
  const __bf16* vbase = Vt + (size_t)(h * HD) * MTOT + b * SEQ + lhi * 8;

  // prime the K-fragment register pipeline
  v16bf ka[4];
  {
    const __bf16* kr0 = kbase + (size_t)lrow * DIMC;
    const __bf16* kr1 = kr0 + (size_t)16 * DIMC;
    ka[0] = load_afrag_bf16(kr0);
    ka[1] = load_afrag_bf16(kr0 + 32);
    ka[2] = load_afrag_bf16(kr1);
    ka[3] = load_afrag_bf16(kr1 + 32);
  }

  for (int kb = 0; kb < SEQ; kb += 32) {
    const int kb2 = (kb + 32 < SEQ) ? kb + 32 : 0;   // branchless wrap for last iter

    // issue next K block + current V block loads before any compute
    v16bf kn[4];
    {
      const __bf16* kr0 = kbase + (size_t)(kb2 + lrow) * DIMC;
      const __bf16* kr1 = kr0 + (size_t)16 * DIMC;
      kn[0] = load_afrag_bf16(kr0);
      kn[1] = load_afrag_bf16(kr0 + 32);
      kn[2] = load_afrag_bf16(kr1);
      kn[3] = load_afrag_bf16(kr1 + 32);
    }
    v16bf va[4];
#pragma unroll
    for (int t = 0; t < 4; ++t)
      va[t] = load_afrag_bf16(vbase + (size_t)(t * 16 + lrow) * MTOT + kb);
    __builtin_prefetch(vbase + (size_t)(lane * 2) * MTOT + kb2, 0, 1);  // next V block

    // --- scores on the previously-fetched K fragments (no load wait here) ---
    v8f s0 = {}, s1 = {};        // S^T tiles: rows keys, cols queries
    s0 = WMMA_BF16(ka[0], qb0, s0);
    s0 = WMMA_BF16(ka[1], qb1, s0);
    s1 = WMMA_BF16(ka[2], qb0, s1);
    s1 = WMMA_BF16(ka[3], qb1, s1);

    // --- scale 1/sqrt(DIM)=1/32; online softmax (keys are in-lane now) ---
    float mb = -1e30f;
#pragma unroll
    for (int i = 0; i < 8; ++i) {
      s0[i] *= 0.03125f;
      s1[i] *= 0.03125f;
      mb = fmaxf(mb, fmaxf(s0[i], s1[i]));
    }
    mb = fmaxf(mb, __shfl_xor(mb, 16, 32));   // join the two key half-ranges
    const float mn    = fmaxf(m, mb);
    const float alpha = __expf(m - mn);
    m = mn;
    float rs = 0.f;
#pragma unroll
    for (int i = 0; i < 8; ++i) {
      s0[i] = __expf(s0[i] - mn);
      s1[i] = __expf(s1[i] - mn);
      rs += s0[i] + s1[i];
    }
    rs += __shfl_xor(rs, 16, 32);
    l = l * alpha + rs;
#pragma unroll
    for (int t = 0; t < 4; ++t)
#pragma unroll
      for (int i = 0; i < 8; ++i) o[t][i] *= alpha;

    // --- P^T -> per-wave LDS [query][key_local]; wave-local ordering only ---
    v8bf p0, p1;
#pragma unroll
    for (int i = 0; i < 8; ++i) { p0[i] = (__bf16)s0[i]; p1[i] = (__bf16)s1[i]; }
    *(v8bf*)&lds_pt[wave][lrow][lhi * 8]      = p0;
    *(v8bf*)&lds_pt[wave][lrow][16 + lhi * 8] = p1;
    wave_lds_fence();
    const v16bf pb = load_bfrag(&lds_pt[wave][lrow][lhi * 8]);

    // --- O^T += V^T @ P^T (V fragments fetched at loop top) ---
#pragma unroll
    for (int t = 0; t < 4; ++t) o[t] = WMMA_BF16(va[t], pb, o[t]);
    wave_lds_fence();            // protect lds_pt against next iteration's store

#pragma unroll
    for (int t = 0; t < 4; ++t) ka[t] = kn[t];
  }

  // --- normalize (scalar per lane) and store context: 4 x 16B contiguous ---
  const float inv = 1.0f / l;
  const size_t crow = (size_t)(b * SEQ + q0 + lrow) * DIMC + h * HD;
#pragma unroll
  for (int t = 0; t < 4; ++t) {
    v8bf ov;
#pragma unroll
    for (int i = 0; i < 8; ++i) ov[i] = (__bf16)(o[t][i] * inv);
    *(v8bf*)(Ctx + crow + t * 16 + lhi * 8) = ov;
  }
}

// ---------------------------------------------------------------------------
extern "C" void kernel_launch(void* const* d_in, const int* in_sizes, int n_in,
                              void* d_out, int out_size, void* d_ws, size_t ws_size,
                              hipStream_t stream) {
  (void)in_sizes; (void)n_in; (void)out_size; (void)ws_size;

  const float* q  = (const float*)d_in[0];
  const float* k  = (const float*)d_in[1];
  const float* v  = (const float*)d_in[2];
  const float* Wq = (const float*)d_in[3];
  const float* bq = (const float*)d_in[4];
  const float* Wk = (const float*)d_in[5];
  const float* bk = (const float*)d_in[6];
  const float* Wv = (const float*)d_in[7];
  const float* bv = (const float*)d_in[8];
  const float* Wo = (const float*)d_in[9];
  const float* bo = (const float*)d_in[10];

  char* ws = (char*)d_ws;
  const size_t MAT = (size_t)MTOT * DIMC * sizeof(__bf16);   // 8 MB each
  __bf16* Qp  = (__bf16*)(ws + 0 * MAT);
  __bf16* Kp  = (__bf16*)(ws + 1 * MAT);
  __bf16* Vt  = (__bf16*)(ws + 2 * MAT);   // transposed [DIM][MTOT]
  __bf16* Ctx = (__bf16*)(ws + 3 * MAT);

  const dim3 blk(256);
  const dim3 gproj(MTOT / 256, DIMC / 64);   // 16 x 16 workgroups

  proj_kernel<float, 0><<<gproj, blk, 0, stream>>>(q, Wq, bq, Qp);
  proj_kernel<float, 0><<<gproj, blk, 0, stream>>>(k, Wk, bk, Kp);
  proj_kernel<float, 1><<<gproj, blk, 0, stream>>>(v, Wv, bv, Vt);
  attn_kernel<<<dim3(SEQ / 128, NB * NH), blk, 0, stream>>>(Qp, Kp, Vt, Ctx);
  proj_kernel<__bf16, 2><<<gproj, blk, 0, stream>>>(Ctx, Wo, bo, d_out);
}
